// ElmoAspectSent_15315853377637
// MI455X (gfx1250) — matrix operand order repaired and checked by reading, not compiled, not run on hardware
//
#include <hip/hip_runtime.h>
#include <hip/hip_bf16.h>

// ---------------------------------------------------------------------------
// Shapes
// ---------------------------------------------------------------------------
static constexpr int kB    = 64;
static constexpr int kT    = 256;
static constexpr int kD    = 1024;
static constexpr int kMD   = 50;
static constexpr int kH    = 256;   // hidden per direction
static constexpr int kHID  = 512;   // 2*kH
static constexpr int kDIN  = 1074;  // kD + kMD
static constexpr int kDINP = 1088;  // padded to multiple of 32
static constexpr int kG4H  = 1024;  // 4*kH (gate width)

typedef __attribute__((ext_vector_type(16))) _Float16 v16h;
typedef __attribute__((ext_vector_type(8)))  float    v8f;

union FragA { v16h h; unsigned u[8]; };

#if defined(__gfx1250__) && __has_builtin(__builtin_amdgcn_global_load_async_to_lds_b128)
#define HAVE_ASYNC_LDS 1
#else
#define HAVE_ASYNC_LDS 0
#endif

#if HAVE_ASYNC_LDS
typedef int v4i_vs __attribute__((vector_size(4 * sizeof(int))));
typedef __attribute__((address_space(1))) v4i_vs* async_gptr;
typedef __attribute__((address_space(3))) v4i_vs* async_lptr;
#endif

// async global -> LDS copy of 16 bytes (falls back to a plain copy)
static __device__ inline void async_copy_b128(const void* gsrc, void* ldst) {
#if HAVE_ASYNC_LDS
  __builtin_amdgcn_global_load_async_to_lds_b128(
      (async_gptr)(void*)gsrc, (async_lptr)ldst, 0, 0);
#else
  *(float4*)ldst = *(const float4*)gsrc;
#endif
}
static __device__ inline void async_wait0() {
#if HAVE_ASYNC_LDS
#if __has_builtin(__builtin_amdgcn_s_wait_asynccnt)
  __builtin_amdgcn_s_wait_asynccnt(0);
#else
  asm volatile("s_wait_asynccnt 0x0" ::: "memory");
#endif
#endif
}

static __device__ inline v8f vzero8() {
  v8f z = {0.f,0.f,0.f,0.f,0.f,0.f,0.f,0.f};
  return z;
}
static __device__ inline v8f wmma_f16(v16h a, v16h b, v8f c) {
  // (neg_a, A, neg_b, B, c_mod, C, reuse_a, reuse_b)
  return __builtin_amdgcn_wmma_f32_16x16x32_f16(false, a, false, b, (short)0, c, false, false);
}
static __device__ inline float sigf(float x) { return 1.f / (1.f + __expf(-x)); }
static __device__ inline float lse2(float x, float y) {
  float m = fmaxf(x, y);
  return m + __logf(__expf(x - m) + __expf(y - m));
}

// ---------------------------------------------------------------------------
// Kernel 1: convert weights to f16 (W_ih padded to kDINP cols, W_hh as-is)
// ---------------------------------------------------------------------------
__global__ void convert_w_kernel(const float* __restrict__ Wf, const float* __restrict__ Wb,
                                 const float* __restrict__ Hf, const float* __restrict__ Hb,
                                 _Float16* __restrict__ WFo, _Float16* __restrict__ WBo,
                                 _Float16* __restrict__ HFo, _Float16* __restrict__ HBo) {
  const int n1 = kG4H * kDINP;
  const int n2 = kG4H * kH;
  for (int i = blockIdx.x * blockDim.x + threadIdx.x; i < n1; i += gridDim.x * blockDim.x) {
    int r = i / kDINP, c = i % kDINP;
    float vf = (c < kDIN) ? Wf[r * kDIN + c] : 0.f;
    float vb = (c < kDIN) ? Wb[r * kDIN + c] : 0.f;
    WFo[i] = (_Float16)vf;
    WBo[i] = (_Float16)vb;
    if (i < n2) {
      HFo[i] = (_Float16)Hf[i];
      HBo[i] = (_Float16)Hb[i];
    }
  }
}

// ---------------------------------------------------------------------------
// Kernel 2: build x = concat(sents, mask_embed[masks]) in f16, padded to kDINP
// ---------------------------------------------------------------------------
__global__ void build_x_kernel(const float* __restrict__ sents, const int* __restrict__ masks,
                               const float* __restrict__ mask_embed, _Float16* __restrict__ XH) {
  const size_t n = (size_t)kB * kT * kDINP;
  for (size_t i = (size_t)blockIdx.x * blockDim.x + threadIdx.x; i < n;
       i += (size_t)gridDim.x * blockDim.x) {
    size_t row = i / kDINP;
    int col = (int)(i % kDINP);
    float v;
    if (col < kD)           v = sents[row * kD + col];
    else if (col < kDIN)    v = mask_embed[masks[row] * kMD + (col - kD)];
    else                    v = 0.f;
    XH[i] = (_Float16)v;
  }
}

// ---------------------------------------------------------------------------
// Kernel 3: G = x @ W_ih^T + b   (f16 x f16 -> f32 -> f16)   via WMMA
//   grid: (M/128, N/64, 2 dirs), block: 256 threads (8 waves)
//   Double-buffered LDS staging via async global->LDS copies.
// ---------------------------------------------------------------------------
__global__ __launch_bounds__(256)
void gemm_xproj_kernel(const _Float16* __restrict__ XH,
                       const _Float16* __restrict__ WF, const _Float16* __restrict__ WB,
                       const float* __restrict__ bF, const float* __restrict__ bB,
                       _Float16* __restrict__ GF, _Float16* __restrict__ GB) {
  const int dir = blockIdx.z;
  const _Float16* W    = dir ? WB : WF;
  const float*    bias = dir ? bB : bF;
  _Float16*       G    = dir ? GB : GF;

  const int row0 = blockIdx.x * 128;
  const int col0 = blockIdx.y * 64;

  __shared__ _Float16 sA[2][128 * 32];   // 16 KB
  __shared__ _Float16 sB[2][64 * 32];    //  8 KB

  const int tid  = threadIdx.x;
  const int lane = tid & 31;
  const int wave = tid >> 5;
  const int mp   = wave & 3;   // m-pair 0..3
  const int np   = wave >> 2;  // n-pair 0..1
  const int l16  = lane & 15;
  const int half = lane >> 4;

  // staging assignments
  const int ar = tid >> 1, as = tid & 1;  // A: row, 16-f16 segment (x2 chunks)
  const int bn = tid >> 2, bs = tid & 3;  // B: row, 8-f16 chunk

  auto stage = [&](int kk, int buf) {
    const _Float16* aA = XH + (size_t)(row0 + ar) * kDINP + kk * 32 + as * 16;
    async_copy_b128(aA,     &sA[buf][ar * 32 + as * 16]);
    async_copy_b128(aA + 8, &sA[buf][ar * 32 + as * 16 + 8]);
    const _Float16* aB = W + (size_t)(col0 + bn) * kDINP + kk * 32 + bs * 8;
    async_copy_b128(aB, &sB[buf][bn * 32 + bs * 8]);
  };

  v8f acc[2][2];
  acc[0][0] = vzero8(); acc[0][1] = vzero8();
  acc[1][0] = vzero8(); acc[1][1] = vzero8();

  stage(0, 0);
  async_wait0();
  __syncthreads();

  const int KK = kDINP / 32;
  for (int kk = 0; kk < KK; ++kk) {
    const int cur = kk & 1;
    if (kk + 1 < KK) stage(kk + 1, cur ^ 1);

    FragA A2[2], B2[2];
#pragma unroll
    for (int mi = 0; mi < 2; ++mi) {
      int row = mp * 32 + mi * 16 + l16;
#pragma unroll
      for (int v = 0; v < 8; ++v) {
        int kbase = (v >> 2) * 16 + half * 8 + (v & 3) * 2;
        A2[mi].u[v] = *(const unsigned*)(&sA[cur][row * 32 + kbase]);
      }
    }
#pragma unroll
    for (int ni = 0; ni < 2; ++ni) {
      int n = np * 32 + ni * 16 + l16;
#pragma unroll
      for (int v = 0; v < 8; ++v) {
        int k = (v >> 2) * 16 + half * 8 + (v & 3) * 2;
        B2[ni].u[v] = *(const unsigned*)(&sB[cur][n * 32 + k]);
      }
    }
#pragma unroll
    for (int mi = 0; mi < 2; ++mi)
#pragma unroll
      for (int ni = 0; ni < 2; ++ni)
        acc[mi][ni] = wmma_f16(A2[mi].h, B2[ni].h, acc[mi][ni]);

    async_wait0();
    __syncthreads();
  }

#pragma unroll
  for (int mi = 0; mi < 2; ++mi)
#pragma unroll
    for (int ni = 0; ni < 2; ++ni) {
      int gcol = col0 + np * 32 + ni * 16 + l16;
      float bv = bias[gcol];
#pragma unroll
      for (int r = 0; r < 8; ++r) {
        int grow = row0 + mp * 32 + mi * 16 + r + 8 * half;
        G[(size_t)grow * kG4H + gcol] = (_Float16)(acc[mi][ni][r] + bv);
      }
    }
}

// ---------------------------------------------------------------------------
// Kernel 4: recurrent LSTM. grid (4 batch-groups, 2 dirs), 1024 threads.
//   per step: gbuf <- async G[b, tsel] slice;  gbuf += h @ W_hh^T (WMMA)
// ---------------------------------------------------------------------------
__global__ __launch_bounds__(1024)
void lstm_kernel(const _Float16* __restrict__ GF, const _Float16* __restrict__ GB,
                 const _Float16* __restrict__ WhhF, const _Float16* __restrict__ WhhB,
                 const int* __restrict__ lens,
                 float* __restrict__ outF, float* __restrict__ outB) {
  const int dir = blockIdx.y;
  const _Float16* G   = dir ? GB   : GF;
  const _Float16* Whh = dir ? WhhB : WhhF;
  float*          Out = dir ? outB : outF;
  const int bg0 = blockIdx.x * 16;

  __shared__ _Float16 hbuf[16 * kH];     // 8 KB  (h, A-operand)
  __shared__ _Float16 gbuf[16 * kG4H];   // 32 KB (input proj + recurrent gates)
  __shared__ int lens_s[16];

  const int tid  = threadIdx.x;
  const int lane = tid & 31;
  const int wave = tid >> 5;
  const int l16  = lane & 15;
  const int half = lane >> 4;
  const int nt0  = wave * 2;             // 32 waves x 2 n-tiles = 64 tiles (N=1024)

  if (tid < 16) lens_s[tid] = lens[bg0 + tid];
  for (int i = tid; i < 16 * kH; i += 1024) hbuf[i] = (_Float16)0.f;

  // async G-slice staging: 16 rows x 1024 f16 = 2048 chunks of 16B, 2/thread
  const int c0   = tid * 2;
  const int gM0  = c0 >> 7;               // both chunks share the same row
  const int gOff = (c0 & 127) * 8;        // f16 offset within the row

  // elementwise ownership: col j = tid%256, rows (tid/256)*4 .. +3
  const int ecol  = tid & 255;
  const int erow0 = (tid >> 8) * 4;
  float creg[4] = {0.f, 0.f, 0.f, 0.f};

  __syncthreads();

  for (int t = 0; t < kT; ++t) {
    // ---- issue async copy of G[b, tsel(b)] slice into gbuf ----
    {
      int len  = lens_s[gM0];
      int tsel = dir ? min(max(len - 1 - t, 0), kT - 1) : t;
      const _Float16* src = G + ((size_t)(bg0 + gM0) * kT + tsel) * kG4H + gOff;
      async_copy_b128(src,     &gbuf[gM0 * kG4H + gOff]);
      async_copy_b128(src + 8, &gbuf[gM0 * kG4H + gOff + 8]);
    }

    // ---- cache A fragments (h) for all 8 k-windows ----
    FragA afr[8];
#pragma unroll
    for (int kk = 0; kk < 8; ++kk)
#pragma unroll
      for (int v = 0; v < 8; ++v) {
        int kbase = kk * 32 + (v >> 2) * 16 + half * 8 + (v & 3) * 2;
        afr[kk].u[v] = *(const unsigned*)(hbuf + l16 * kH + kbase);
      }

    // ---- recurrent GEMM: 2 n-tiles per wave ----
    v8f accs[2];
#pragma unroll
    for (int ni = 0; ni < 2; ++ni) {
      const int n = (nt0 + ni) * 16 + l16;
      v8f acc = vzero8();
#pragma unroll
      for (int kk = 0; kk < 8; ++kk) {
        FragA bfr;
#pragma unroll
        for (int v = 0; v < 8; ++v) {
          int k = kk * 32 + (v >> 2) * 16 + half * 8 + (v & 3) * 2;
          bfr.u[v] = *(const unsigned*)(Whh + (size_t)n * kH + k);
        }
        acc = wmma_f16(afr[kk].h, bfr.h, acc);
      }
      accs[ni] = acc;
    }

    async_wait0();
    __syncthreads();   // gbuf holds the input-projection slice

    // ---- epilogue: gbuf += recurrent part (LDS RMW, disjoint ownership) ----
#pragma unroll
    for (int ni = 0; ni < 2; ++ni) {
      const int n = (nt0 + ni) * 16 + l16;
#pragma unroll
      for (int r = 0; r < 8; ++r) {
        int M = r + 8 * half;
        float g = (float)gbuf[M * kG4H + n] + accs[ni][r];
        gbuf[M * kG4H + n] = (_Float16)g;
      }
    }
    __syncthreads();

    // ---- elementwise gate math ----
#pragma unroll
    for (int rr = 0; rr < 4; ++rr) {
      const int r = erow0 + rr;
      const int len = lens_s[r];
      if (t < len) {
        float gi = (float)gbuf[r * kG4H + ecol];
        float gf = (float)gbuf[r * kG4H + kH + ecol];
        float gg = (float)gbuf[r * kG4H + 2 * kH + ecol];
        float go = (float)gbuf[r * kG4H + 3 * kH + ecol];
        float cn = sigf(gf) * creg[rr] + sigf(gi) * tanhf(gg);
        float hn = sigf(go) * tanhf(cn);
        creg[rr] = cn;
        hbuf[r * kH + ecol] = (_Float16)hn;
        int tpos = dir ? (len - 1 - t) : t;
        Out[((size_t)(bg0 + r) * kT + tpos) * kH + ecol] = hn;
      }
    }
    __syncthreads();
  }
}

// ---------------------------------------------------------------------------
// context accessor helpers (context = [out_f, out_b], HID=512)
// ---------------------------------------------------------------------------
static __device__ inline float ctx_at(const float* outF, const float* outB,
                                      int b, int t, int h) {
  return (h < kH) ? outF[((size_t)b * kT + t) * kH + h]
                  : outB[((size_t)b * kT + t) * kH + (h - kH)];
}

// Kernel 5: avg over masked positions
__global__ void avg_kernel(const float* outF, const float* outB, const int* masks,
                           float* __restrict__ AVG) {
  int i = blockIdx.x * blockDim.x + threadIdx.x;
  if (i >= kB * kHID) return;
  int b = i / kHID, h = i % kHID;
  float s = 0.f, cnt = 0.f;
  for (int t = 0; t < kT; ++t) {
    float m = (float)masks[b * kT + t];
    s += m * ctx_at(outF, outB, b, t, h);
    cnt += m;
  }
  AVG[i] = s / cnt;
}

// Kernel 6: tri = (context + avg) @ W_tri^T + b_tri
__global__ void tri_kernel(const float* outF, const float* outB, const float* AVG,
                           const float* __restrict__ W_tri, const float* __restrict__ b_tri,
                           float* __restrict__ TRI) {
  int i = blockIdx.x * blockDim.x + threadIdx.x;
  if (i >= kB * kT) return;
  int b = i / kT, t = i % kT;
  float s0 = b_tri[0], s1 = b_tri[1];
  for (int h = 0; h < kHID; ++h) {
    float x = ctx_at(outF, outB, b, t, h) + AVG[b * kHID + h];
    s0 += x * W_tri[h];
    s1 += x * W_tri[kHID + h];
  }
  TRI[(size_t)i * 2]     = s0;
  TRI[(size_t)i * 2 + 1] = s1;
}

// Kernel 7: 2-state CRF forward/backward marginals -> SEL = marg[:, :, 1]
__global__ void crf_kernel(const float* __restrict__ TRI, const int* __restrict__ lens,
                           const float* __restrict__ trans,
                           float* __restrict__ ALPHA, float* __restrict__ BETA,
                           float* __restrict__ SEL) {
  int b = blockIdx.x * blockDim.x + threadIdx.x;
  if (b >= kB) return;
  float t00 = trans[0], t01 = trans[1], t10 = trans[2], t11 = trans[3];
  int L = lens[b];
  const float* f = TRI + (size_t)b * kT * 2;
  float* A = ALPHA + (size_t)b * kT * 2;
  float* Bv = BETA + (size_t)b * kT * 2;

  float a0 = f[0], a1 = f[1];
  A[0] = a0; A[1] = a1;
  for (int t = 1; t < kT; ++t) {
    float c0 = f[t * 2 + 0] + lse2(a0 + t00, a1 + t01);
    float c1 = f[t * 2 + 1] + lse2(a0 + t10, a1 + t11);
    if (t < L) { a0 = c0; a1 = c1; }
    A[t * 2] = a0; A[t * 2 + 1] = a1;
  }
  float lz = lse2(A[(L - 1) * 2], A[(L - 1) * 2 + 1]);

  float b0 = 0.f, b1 = 0.f;
  Bv[(kT - 1) * 2] = 0.f; Bv[(kT - 1) * 2 + 1] = 0.f;
  for (int t = kT - 2; t >= 0; --t) {
    float fn0 = f[(t + 1) * 2], fn1 = f[(t + 1) * 2 + 1];
    float c0 = lse2(b0 + fn0 + t00, b1 + fn1 + t10);
    float c1 = lse2(b0 + fn0 + t01, b1 + fn1 + t11);
    if (t < L - 1) { b0 = c0; b1 = c1; } else { b0 = 0.f; b1 = 0.f; }
    Bv[t * 2] = b0; Bv[t * 2 + 1] = b1;
  }
  for (int t = 0; t < kT; ++t)
    SEL[b * kT + t] = (t < L) ? __expf(A[t * 2 + 1] + Bv[t * 2 + 1] - lz) : 0.f;
}

// Kernel 8: sent_v = einsum('bt,bth->bh', sel, context)
__global__ void sentv_kernel(const float* outF, const float* outB, const float* SEL,
                             float* __restrict__ SENTV) {
  int i = blockIdx.x * blockDim.x + threadIdx.x;
  if (i >= kB * kHID) return;
  int b = i / kHID, h = i % kHID;
  float s = 0.f;
  for (int t = 0; t < kT; ++t) s += SEL[b * kT + t] * ctx_at(outF, outB, b, t, h);
  SENTV[i] = s;
}

// Kernel 9: classifier loss + penalties -> out[2]
__global__ void final_kernel(const float* __restrict__ SENTV,
                             const float* __restrict__ W_lab, const float* __restrict__ b_lab,
                             const int* __restrict__ labels, const float* __restrict__ SEL,
                             const float* __restrict__ trans, float* __restrict__ out) {
  __shared__ float rloss[kB];
  __shared__ float rnorm[kB];
  int b = threadIdx.x;
  if (b < kB) {
    float s[3];
    for (int k = 0; k < 3; ++k) {
      float a = b_lab[k];
      for (int h = 0; h < kHID; ++h) a += SENTV[b * kHID + h] * W_lab[k * kHID + h];
      s[k] = a;
    }
    float mx = fmaxf(s[0], fmaxf(s[1], s[2]));
    float lse = mx + __logf(__expf(s[0] - mx) + __expf(s[1] - mx) + __expf(s[2] - mx));
    rloss[b] = -(s[labels[b]] - lse);
    float sn = 0.f;
    for (int t = 0; t < kT; ++t) sn += fabsf(SEL[b * kT + t]);
    rnorm[b] = sn;
  }
  __syncthreads();
  if (b == 0) {
    float L = 0.f, N = 0.f;
    for (int i = 0; i < kB; ++i) { L += rloss[i]; N += rnorm[i]; }
    float pena = fmaxf(trans[2] - trans[0], 0.f) + fmaxf(trans[1] - trans[3], 0.f);
    out[0] = L / (float)kB;
    out[1] = pena / (float)kB + N / (float)kB;   // C1 = C2 = 1
  }
}

// ---------------------------------------------------------------------------
// Host launcher
// ---------------------------------------------------------------------------
extern "C" void kernel_launch(void* const* d_in, const int* in_sizes, int n_in,
                              void* d_out, int out_size, void* d_ws, size_t ws_size,
                              hipStream_t stream) {
  const float* sents      = (const float*)d_in[0];
  const int*   masks      = (const int*)d_in[1];
  const int*   lens       = (const int*)d_in[2];
  const int*   labels     = (const int*)d_in[3];
  const float* mask_embed = (const float*)d_in[4];
  const float* W_ih_f     = (const float*)d_in[5];
  const float* W_hh_f     = (const float*)d_in[6];
  const float* b_f        = (const float*)d_in[7];
  const float* W_ih_b     = (const float*)d_in[8];
  const float* W_hh_b     = (const float*)d_in[9];
  const float* b_b        = (const float*)d_in[10];
  const float* W_tri      = (const float*)d_in[11];
  const float* b_tri      = (const float*)d_in[12];
  const float* W_lab      = (const float*)d_in[13];
  const float* b_lab      = (const float*)d_in[14];
  const float* trans      = (const float*)d_in[15];
  float* out = (float*)d_out;

  // workspace carve-up
  char* base = (char*)d_ws;
  size_t off = 0;
  auto carve = [&](size_t bytes) -> char* {
    char* p = base + off;
    off += (bytes + 255) & ~(size_t)255;
    return p;
  };
  _Float16* XH   = (_Float16*)carve((size_t)kB * kT * kDINP * 2);
  _Float16* WIHF = (_Float16*)carve((size_t)kG4H * kDINP * 2);
  _Float16* WIHB = (_Float16*)carve((size_t)kG4H * kDINP * 2);
  _Float16* WHHF = (_Float16*)carve((size_t)kG4H * kH * 2);
  _Float16* WHHB = (_Float16*)carve((size_t)kG4H * kH * 2);
  _Float16* GF   = (_Float16*)carve((size_t)kB * kT * kG4H * 2);
  _Float16* GB   = (_Float16*)carve((size_t)kB * kT * kG4H * 2);
  float* OUTF  = (float*)carve((size_t)kB * kT * kH * 4);
  float* OUTB  = (float*)carve((size_t)kB * kT * kH * 4);
  float* AVG   = (float*)carve((size_t)kB * kHID * 4);
  float* TRI   = (float*)carve((size_t)kB * kT * 2 * 4);
  float* ALPHA = (float*)carve((size_t)kB * kT * 2 * 4);
  float* BETA  = (float*)carve((size_t)kB * kT * 2 * 4);
  float* SEL   = (float*)carve((size_t)kB * kT * 4);
  float* SENTV = (float*)carve((size_t)kB * kHID * 4);

  // zero-init masked outputs
  (void)hipMemsetAsync(OUTF, 0, (size_t)kB * kT * kH * 4, stream);
  (void)hipMemsetAsync(OUTB, 0, (size_t)kB * kT * kH * 4, stream);

  convert_w_kernel<<<2048, 256, 0, stream>>>(W_ih_f, W_ih_b, W_hh_f, W_hh_b,
                                             WIHF, WIHB, WHHF, WHHB);
  build_x_kernel<<<4096, 256, 0, stream>>>(sents, masks, mask_embed, XH);

  dim3 ggrid(kB * kT / 128, kG4H / 64, 2);
  gemm_xproj_kernel<<<ggrid, 256, 0, stream>>>(XH, WIHF, WIHB, b_f, b_b, GF, GB);

  dim3 lgrid(4, 2);
  lstm_kernel<<<lgrid, 1024, 0, stream>>>(GF, GB, WHHF, WHHB, lens, OUTF, OUTB);

  avg_kernel<<<(kB * kHID + 255) / 256, 256, 0, stream>>>(OUTF, OUTB, masks, AVG);
  tri_kernel<<<(kB * kT + 255) / 256, 256, 0, stream>>>(OUTF, OUTB, AVG, W_tri, b_tri, TRI);
  crf_kernel<<<1, 64, 0, stream>>>(TRI, lens, trans, ALPHA, BETA, SEL);
  sentv_kernel<<<(kB * kHID + 255) / 256, 256, 0, stream>>>(OUTF, OUTB, SEL, SENTV);
  final_kernel<<<1, 64, 0, stream>>>(SENTV, W_lab, b_lab, labels, SEL, trans, out);
}
